// GPTQLinearCPU_27436251087100
// MI455X (gfx1250) — matrix operand order repaired and checked by reading, not compiled, not run on hardware
//
#include <hip/hip_runtime.h>

typedef _Float16 h16;
typedef __attribute__((ext_vector_type(16))) _Float16 v16h;
typedef __attribute__((ext_vector_type(8)))  _Float16 v8h;
typedef __attribute__((ext_vector_type(2)))  _Float16 v2h;
typedef __attribute__((ext_vector_type(8)))  float    v8f;
typedef __attribute__((ext_vector_type(4)))  unsigned u32x4;

#define BM 128
#define BN 128
#define BK 32
#define LDT 40               // padded row stride (halves) for A and B LDS tiles
#define TILE_H (128 * LDT)   // halves per tile (A or B)

// GEMM: out[M,OUT] = x[M,IN] (fp16) * dequant(qweight)[IN,OUT] + bias
// qweight[IN/8, OUT] int32, 8 int4 per word along IN; scale per 128 IN rows.
__global__ __launch_bounds__(256)
void gptq_wmma_kernel(const h16* __restrict__ x,
                      const int* __restrict__ qw,
                      const h16* __restrict__ scales,
                      const h16* __restrict__ bias,
                      float* __restrict__ out,
                      int M, int IN, int OUT)
{
    const int t    = threadIdx.x;
    const int lane = t & 31;
    const int wave = t >> 5;
    const int sel  = lane >> 4;   // half-wave select (0: lanes 0-15, 1: lanes 16-31)
    const int ln   = lane & 15;

    const int gM = blockIdx.y * BM;
    const int gN = blockIdx.x * BN;

    // 8 waves arranged 2 (M) x 4 (N); each wave owns a 64x32 C subtile
    const int waveM = (wave & 1) * 64;
    const int waveN = (wave >> 1) * 32;

    __shared__ h16 smem[4 * TILE_H];  // [buf][A|B][128][LDT]

    const int KT = IN / BK;

    // accumulators: 4 (M) x 2 (N) fragments of 16x16 f32
    v8f acc[4][2];
    v8f zero;
    #pragma unroll
    for (int i = 0; i < 8; ++i) zero[i] = 0.0f;
    #pragma unroll
    for (int mi = 0; mi < 4; ++mi)
        #pragma unroll
        for (int ni = 0; ni < 2; ++ni) acc[mi][ni] = zero;

    // ---- global-load staging (register double buffer) ----
    v8h xa[2][2];   // x tile: 128 rows x 32 halves; thread covers 2 rows x 16 halves
    int qv[2];      // 2 packed int32 (8 int4 each)
    h16 sv;         // group scale for this thread's output column

    // A-tile indices: chunk id c = t -> row c>>2 (0..63 and +64), 8-half chunk c&3
    const int arow = t >> 2;
    const int ach  = t & 3;
    // B-tile indices: 512 int32 = 4 packed-k rows x 128 cols; thread does pr and pr+2
    const int bn = t & 127;
    const int bp = t >> 7;   // 0..1

    auto load_global = [&](int kt) {
        const h16* xp = x + (size_t)(gM + arow) * IN + (size_t)kt * BK + ach * 8;
        xa[0][0] = *(const v8h*)xp;
        xa[0][1] = *(const v8h*)(xp + 8);
        const h16* xp2 = xp + (size_t)64 * IN;
        xa[1][0] = *(const v8h*)xp2;
        xa[1][1] = *(const v8h*)(xp2 + 8);
        const int* qp = qw + (size_t)(kt * 4 + bp) * OUT + gN + bn;
        qv[0] = qp[0];
        qv[1] = qp[(size_t)2 * OUT];
        sv = scales[(size_t)(kt >> 2) * OUT + gN + bn];  // 32-K tile within 128-K group
    };

    // dequant one packed fp16 pair with fused (nib-8)*s:
    //   fp16(0x6400 | nib) == 1024 + nib  ->  (nib-8)*s = val*s + (-1032*s)
    auto dq_pair = [](unsigned pw, v2h s2, v2h k2) -> unsigned {
        union { unsigned u; v2h h; } a, r;
        a.u = pw;
        r.h = __builtin_elementwise_fma(a.h, s2, k2);
        return r.u;
    };

    auto store_lds = [&](int buf) {
        h16* A = smem + buf * 2 * TILE_H;
        h16* B = A + TILE_H;                       // B stored N-major: [n][k], k=0..31
        // x tile: straight copy, 16B stores
        {
            h16* ap = A + arow * LDT + ach * 8;
            *(v8h*)ap = xa[0][0];
            *(v8h*)(ap + 8) = xa[0][1];
            h16* ap2 = A + (arow + 64) * LDT + ach * 8;
            *(v8h*)ap2 = xa[1][0];
            *(v8h*)(ap2 + 8) = xa[1][1];
        }
        const h16 nks = (h16)(-1032.0f) * sv;
        v2h s2, k2;
        s2[0] = sv;  s2[1] = sv;
        k2[0] = nks; k2[1] = nks;
        #pragma unroll
        for (int r = 0; r < 2; ++r) {
            const int pr = bp + r * 2;             // packed row in tile (k0 = pr*8)
            const unsigned q = (unsigned)qv[r];
            // interleaved pairs: r_j = {k_j , k_{j+4}} dequantized via packed FMA
            unsigned r0 = dq_pair(( q        & 0x000F000Fu) | 0x64006400u, s2, k2);
            unsigned r1 = dq_pair(((q >>  4) & 0x000F000Fu) | 0x64006400u, s2, k2);
            unsigned r2 = dq_pair(((q >>  8) & 0x000F000Fu) | 0x64006400u, s2, k2);
            unsigned r3 = dq_pair(((q >> 12) & 0x000F000Fu) | 0x64006400u, s2, k2);
            // byte-gather to k-order with v_perm_b32 (sel 0-3 -> 2nd arg bytes)
            u32x4 w;
            w[0] = __builtin_amdgcn_perm(r1, r0, 0x05040100u);  // {k0,k1}
            w[1] = __builtin_amdgcn_perm(r3, r2, 0x05040100u);  // {k2,k3}
            w[2] = __builtin_amdgcn_perm(r1, r0, 0x07060302u);  // {k4,k5}
            w[3] = __builtin_amdgcn_perm(r3, r2, 0x07060302u);  // {k6,k7}
            *(u32x4*)(B + bn * LDT + pr * 8) = w;
        }
    };

    load_global(0);
    store_lds(0);

    // Single loop body: one `compute` call site keeps the accumulators pinned
    // in one register set (manual unroll caused full-acc v_mov_b64 copies).
    for (int kt = 0; kt < KT; ++kt) {
        __syncthreads();
        const int buf = kt & 1;
        const bool more = (kt + 1) < KT;
        if (more) load_global(kt + 1);   // overlap global fetch with WMMA below

        const h16* A = smem + buf * 2 * TILE_H;
        const h16* B = A + TILE_H;

        // A fragments per documented 16-bit A layout:
        // lanes 0-15: h[0..7]=K0-7, h[8..15]=K16-23 ; lanes 16-31: K8-15, K24-31
        v16h af[4];
        #pragma unroll
        for (int mi = 0; mi < 4; ++mi) {
            const h16* p = A + (waveM + mi * 16 + ln) * LDT + sel * 8;
            v8h lo = *(const v8h*)p;
            v8h hi = *(const v8h*)(p + 16);
            af[mi] = __builtin_shufflevector(lo, hi,
                0, 1, 2, 3, 4, 5, 6, 7, 8, 9, 10, 11, 12, 13, 14, 15);
        }
        // B fragments: lanes 0-15 hold K0-7, K16-23; lanes 16-31 hold K8-15, K24-31, column N=ln
        v16h bf[2];
        #pragma unroll
        for (int ni = 0; ni < 2; ++ni) {
            const h16* p = B + (waveN + ni * 16 + ln) * LDT + sel * 8;
            v8h lo = *(const v8h*)p;
            v8h hi = *(const v8h*)(p + 16);
            bf[ni] = __builtin_shufflevector(lo, hi,
                0, 1, 2, 3, 4, 5, 6, 7, 8, 9, 10, 11, 12, 13, 14, 15);
        }

        #pragma unroll
        for (int mi = 0; mi < 4; ++mi)
            #pragma unroll
            for (int ni = 0; ni < 2; ++ni)
                acc[mi][ni] = __builtin_amdgcn_wmma_f32_16x16x32_f16(
                    false, af[mi], false, bf[ni],
                    (short)0, acc[mi][ni], false, false);

        if (more) store_lds(buf ^ 1);    // write next tile; readers of `buf` unaffected
    }

    // Epilogue: D layout -> VGPR i: M = i + sel*8, N = ln ; add bias, store fp32
    #pragma unroll
    for (int ni = 0; ni < 2; ++ni) {
        const int col = gN + waveN + ni * 16 + ln;
        const float bz = (float)bias[col];
        #pragma unroll
        for (int mi = 0; mi < 4; ++mi) {
            const int rbase = gM + waveM + mi * 16 + sel * 8;
            #pragma unroll
            for (int i = 0; i < 8; ++i)
                out[(size_t)(rbase + i) * OUT + col] = acc[mi][ni][i] + bz;
        }
    }
}

extern "C" void kernel_launch(void* const* d_in, const int* in_sizes, int n_in,
                              void* d_out, int out_size, void* d_ws, size_t ws_size,
                              hipStream_t stream) {
    const h16* x      = (const h16*)d_in[0];
    const int* qwt    = (const int*)d_in[1];
    const h16* scales = (const h16*)d_in[2];
    const h16* bias   = (const h16*)d_in[3];
    float* out        = (float*)d_out;

    const long long OUT = in_sizes[3];                        // bias length
    const long long IN  = ((long long)in_sizes[1] * 8) / OUT; // qweight = IN/8 * OUT
    const long long M   = (long long)in_sizes[0] / IN;        // x = M * IN

    dim3 grid((unsigned)((OUT + BN - 1) / BN), (unsigned)((M + BM - 1) / BM));
    gptq_wmma_kernel<<<grid, 256, 0, stream>>>(x, qwt, scales, bias, out,
                                               (int)M, (int)IN, (int)OUT);
}
